// NodeGAT_GCN_35141422415878
// MI455X (gfx1250) — matrix operand order repaired and checked by reading, not compiled, not run on hardware
//
#include <hip/hip_runtime.h>
#include <hip/hip_bf16.h>

#define N_NODES 10000
#define E_EDGES 160000
#define ET (E_EDGES + N_NODES)   // with self loops
#define FDIM 128
#define HEADS 10
#define HF (HEADS * FDIM)        // 1280
#define OUTF 1024

typedef __attribute__((ext_vector_type(16))) __bf16 v16bf;
typedef __attribute__((ext_vector_type(8)))  __bf16 v8bf;
typedef __attribute__((ext_vector_type(8)))  float  v8f;
typedef __attribute__((ext_vector_type(4)))  int    v4i;

#if defined(__HIP_DEVICE_COMPILE__) && __has_builtin(__builtin_amdgcn_global_load_async_to_lds_b128)
#define HAVE_ASYNC_LDS 1
typedef __attribute__((address_space(1))) v4i v4i_g;   // global (AS1) int4
typedef __attribute__((address_space(3))) v4i v4i_l;   // LDS (AS3) int4
#else
#define HAVE_ASYNC_LDS 0
#endif

__device__ inline void wait_async0() {
#if HAVE_ASYNC_LDS
#if __has_builtin(__builtin_amdgcn_s_wait_asynccnt)
  __builtin_amdgcn_s_wait_asynccnt(0);
#else
  asm volatile("s_wait_asynccnt 0x0" ::: "memory");
#endif
#endif
}

__device__ inline __bf16 f2bf(float f) { return (__bf16)f; }

// ---------------- elementwise / pack kernels ----------------

__global__ void fill_f32(float* __restrict__ p, float v, int n) {
  int i = blockIdx.x * blockDim.x + threadIdx.x;
  if (i < n) p[i] = v;
}

__global__ void cvt_f32_bf16(const float* __restrict__ in, __bf16* __restrict__ out, int n) {
  int i = blockIdx.x * blockDim.x + threadIdx.x;
  if (i < n) out[i] = f2bf(in[i]);
}

// in[K,N] fp32 row-major -> out[N,K] bf16 row-major (weight pre-transpose)
__global__ void transpose_f32_bf16(const float* __restrict__ in, __bf16* __restrict__ out,
                                   int K, int N) {
  int i = blockIdx.x * blockDim.x + threadIdx.x;
  if (i < K * N) {
    int k = i / N, n = i - k * N;
    out[(size_t)n * K + k] = f2bf(in[i]);
  }
}

__global__ void bias_relu_bf16(const float* __restrict__ in, const float* __restrict__ bias,
                               __bf16* __restrict__ out, int total, int C) {
  int i = blockIdx.x * blockDim.x + threadIdx.x;
  if (i < total) {
    float v = in[i] + bias[i % C];
    out[i] = f2bf(v > 0.f ? v : 0.f);
  }
}

// ---------------- bf16 WMMA GEMM with async-LDS B staging ----------------
// C[M,N](f32) = A[M,K](bf16 row major) x Bt[N,K](bf16 row major == B^T) (+bias)
// block = 160 threads (5 waves), tile = 80(M) x 64(N); wave w -> rows [by*80+w*16, +16)
// B tile (64 cols x 32 K = 4KB) double-buffered in LDS via GLOBAL_LOAD_ASYNC_TO_LDS_B128.
__global__ void __launch_bounds__(160)
gemm_bf16_wmma(const __bf16* __restrict__ A, const __bf16* __restrict__ Bt,
               float* __restrict__ C, const float* __restrict__ bias,
               int M, int N, int K) {
  const int tid  = threadIdx.x;
  const int lane = tid & 31;
  const int wave = tid >> 5;                 // 0..4
  const int m0   = blockIdx.y * 80 + wave * 16;
  const int n0   = blockIdx.x * 64;
  const int lr   = lane & 15;                // row-in-tile (A) / col-in-tile (B,C)
  const int hi   = lane >> 4;                // half-wave selector
  const int kbA  = hi * 16;                  // A frag K offsets (contiguous K 16h..16h+15)

  __shared__ __bf16 smem[2 * 64 * 32];       // double-buffered B tile (2 x 4KB)

  const __bf16* ap = A + (size_t)(m0 + lr) * K + kbA;

  // stage one 64x32 B tile into LDS buffer `buf` (256 chunks of 16B)
  auto stage = [&](int buf, int k) {
    int c0 = tid;
    {
      int col = c0 >> 2, seg = c0 & 3;
      const __bf16* src = Bt + (size_t)(n0 + col) * K + k + seg * 8;
      __bf16* dst = smem + buf * 2048 + c0 * 8;
#if HAVE_ASYNC_LDS
      __builtin_amdgcn_global_load_async_to_lds_b128((v4i_g*)src, (v4i_l*)dst, 0, 0);
#else
      *(v8bf*)dst = *(const v8bf*)src;
#endif
    }
    if (tid < 96) {
      int c1 = tid + 160;
      int col = c1 >> 2, seg = c1 & 3;
      const __bf16* src = Bt + (size_t)(n0 + col) * K + k + seg * 8;
      __bf16* dst = smem + buf * 2048 + c1 * 8;
#if HAVE_ASYNC_LDS
      __builtin_amdgcn_global_load_async_to_lds_b128((v4i_g*)src, (v4i_l*)dst, 0, 0);
#else
      *(v8bf*)dst = *(const v8bf*)src;
#endif
    }
  };

  v8f acc[4];
  acc[0] = (v8f){}; acc[1] = (v8f){}; acc[2] = (v8f){}; acc[3] = (v8f){};

  const int KT = K >> 5;
  stage(0, 0);

  for (int kt = 0; kt < KT; ++kt) {
    wait_async0();          // my async copies landed in LDS
    __syncthreads();        // everyone's copies landed; prior-tile readers done
    if (kt + 1 < KT) {
      stage((kt + 1) & 1, (kt + 1) << 5);
      __builtin_prefetch(ap + ((kt + 1) << 5), 0, 0);   // prefetch next A slab
    }
    const int k = kt << 5;

    // A fragment: two 16B global loads, concat (K kbA..kbA+7, kbA+8..kbA+15)
    const v8bf* pa = reinterpret_cast<const v8bf*>(ap + k);
    v8bf alo = pa[0];
    v8bf ahi = pa[1];
    v16bf afrag = __builtin_shufflevector(alo, ahi,
        0, 1, 2, 3, 4, 5, 6, 7, 8, 9, 10, 11, 12, 13, 14, 15);

    const __bf16* bbase = smem + (kt & 1) * 2048;

    // load all 4 B fragments first so one dscnt wait covers the whole WMMA chain
    v16bf bfrag[4];
#pragma unroll
    for (int j = 0; j < 4; ++j)
      bfrag[j] = *reinterpret_cast<const v16bf*>(bbase + (j * 16 + lr) * 32 + hi * 16);

#pragma unroll
    for (int j = 0; j < 4; ++j)
      acc[j] = __builtin_amdgcn_wmma_f32_16x16x32_bf16(
          false, afrag, false, bfrag[j], (short)0, acc[j], false, false);
  }

#pragma unroll
  for (int j = 0; j < 4; ++j) {
    int n = n0 + j * 16 + lr;
    float bv = bias ? bias[n] : 0.f;
#pragma unroll
    for (int r = 0; r < 8; ++r) {
      int m = m0 + r + hi * 8;
      C[(size_t)m * N + n] = acc[j][r] + bv;
    }
  }
}

// ---------------- GAT attention ----------------

// one wave per (node, head): a_s[n,h] = <x[n,h,:], att_src[h,:]>, same for dst
__global__ void attn_scores(const float* __restrict__ x, const float* __restrict__ att_src,
                            const float* __restrict__ att_dst,
                            float* __restrict__ a_s, float* __restrict__ a_d) {
  int wg = blockIdx.x * (blockDim.x >> 5) + (threadIdx.x >> 5);
  int lane = threadIdx.x & 31;
  if (wg >= N_NODES * HEADS) return;
  int n = wg / HEADS, h = wg - n * HEADS;
  const float* xr = x + (size_t)n * HF + h * FDIM;
  const float* as = att_src + h * FDIM;
  const float* ad = att_dst + h * FDIM;
  float ss = 0.f, sd = 0.f;
#pragma unroll
  for (int i = 0; i < 4; ++i) {
    float xv = xr[lane + 32 * i];
    ss += xv * as[lane + 32 * i];
    sd += xv * ad[lane + 32 * i];
  }
  for (int o = 16; o > 0; o >>= 1) {
    ss += __shfl_down(ss, o, 32);
    sd += __shfl_down(sd, o, 32);
  }
  if (lane == 0) { a_s[wg] = ss; a_d[wg] = sd; }
}

__device__ inline void edge_nodes(const int* __restrict__ ei, int e, int& s, int& d) {
  if (e < E_EDGES) { s = ei[e]; d = ei[E_EDGES + e]; }
  else             { s = e - E_EDGES; d = s; }   // self loop
}

__device__ inline void atomicMaxF32(float* addr, float val) {
  if (val >= 0.f) atomicMax((int*)addr, __float_as_int(val));
  else            atomicMin((unsigned int*)addr, __float_as_uint(val));
}

// per (edge, head): e = leaky_relu(a_s[src]+a_d[dst]); segment max over dst
__global__ void edge_scores(const int* __restrict__ ei, const float* __restrict__ a_s,
                            const float* __restrict__ a_d, float* __restrict__ ev,
                            float* __restrict__ mmax) {
  int t = blockIdx.x * blockDim.x + threadIdx.x;
  if (t >= ET * HEADS) return;
  int e = t / HEADS, h = t - e * HEADS;
  int s, d; edge_nodes(ei, e, s, d);
  float v = a_s[s * HEADS + h] + a_d[d * HEADS + h];
  v = v > 0.f ? v : 0.2f * v;
  ev[t] = v;
  atomicMaxF32(&mmax[d * HEADS + h], v);
}

// per (edge, head): ex = exp(e - m[dst]); segment sum of ex into denom
__global__ void edge_exp(const int* __restrict__ ei, float* __restrict__ ev,
                         const float* __restrict__ mmax, float* __restrict__ denom) {
  int t = blockIdx.x * blockDim.x + threadIdx.x;
  if (t >= ET * HEADS) return;
  int e = t / HEADS, h = t - e * HEADS;
  int s, d; edge_nodes(ei, e, s, d);
  float ex = __expf(ev[t] - mmax[d * HEADS + h]);
  ev[t] = ex;
  atomicAdd(&denom[d * HEADS + h], ex);
}

// block(128) per edge: agg[dst] += alpha * x[src]; also deg[dst] += 1
__global__ void edge_agg(const int* __restrict__ ei, const float* __restrict__ ex,
                         const float* __restrict__ denom, const float* __restrict__ x,
                         float* __restrict__ agg, float* __restrict__ deg) {
  int e = blockIdx.x;
  int s, d; edge_nodes(ei, e, s, d);
  __shared__ float coeff[HEADS];
  if (threadIdx.x < HEADS)
    coeff[threadIdx.x] = ex[e * HEADS + threadIdx.x] /
                         (denom[d * HEADS + threadIdx.x] + 1e-16f);
  if (threadIdx.x == 0) atomicAdd(&deg[d], 1.0f);
  __syncthreads();
  int f = threadIdx.x;
  const float* xs = x + (size_t)s * HF;
  float* ag = agg + (size_t)d * HF;
#pragma unroll
  for (int h = 0; h < HEADS; ++h)
    atomicAdd(&ag[h * FDIM + f], coeff[h] * xs[h * FDIM + f]);
}

// ---------------- GCN ----------------

__global__ void dinv_k(const float* __restrict__ deg, float* __restrict__ dinv) {
  int i = blockIdx.x * blockDim.x + threadIdx.x;
  if (i < N_NODES) {
    float dd = deg[i];
    dinv[i] = dd > 0.f ? rsqrtf(dd) : 0.f;
  }
}

// block(128) per edge: g[dst] += dinv[src]*dinv[dst] * xw[src]
__global__ void gcn_agg(const int* __restrict__ ei, const float* __restrict__ dinv,
                        const float* __restrict__ xw, float* __restrict__ g) {
  int e = blockIdx.x;
  int s, d; edge_nodes(ei, e, s, d);
  float norm = dinv[s] * dinv[d];
  int c = threadIdx.x;
  const float* xs = xw + (size_t)s * HF;
  float* gd = g + (size_t)d * HF;
#pragma unroll
  for (int h = 0; h < HEADS; ++h)
    atomicAdd(&gd[h * FDIM + c], norm * xs[h * FDIM + c]);
}

// ---------------- launcher ----------------

extern "C" void kernel_launch(void* const* d_in, const int* in_sizes, int n_in,
                              void* d_out, int out_size, void* d_ws, size_t ws_size,
                              hipStream_t stream) {
  (void)in_sizes; (void)n_in; (void)out_size; (void)ws_size;
  const int*   edge_index = (const int*)d_in[0];
  const float* node    = (const float*)d_in[1];
  const float* W_gat   = (const float*)d_in[2];
  const float* att_src = (const float*)d_in[3];
  const float* att_dst = (const float*)d_in[4];
  const float* b_gat   = (const float*)d_in[5];
  const float* W_gcn   = (const float*)d_in[6];
  const float* b_gcn   = (const float*)d_in[7];
  const float* W_fc    = (const float*)d_in[8];
  const float* b_fc    = (const float*)d_in[9];
  float* out = (float*)d_out;

  char* base = (char*)d_ws;
  size_t off = 0;
  auto alloc = [&](size_t bytes) -> char* {
    char* p = base + off;
    off += (bytes + 255) & ~(size_t)255;
    return p;
  };
  __bf16* node_bf = (__bf16*)alloc((size_t)N_NODES * FDIM * 2);
  __bf16* wgat_t  = (__bf16*)alloc((size_t)HF * FDIM * 2);
  float*  x       = (float*) alloc((size_t)N_NODES * HF * 4);   // later reused as xw
  float*  a_s     = (float*) alloc((size_t)N_NODES * HEADS * 4);
  float*  a_d     = (float*) alloc((size_t)N_NODES * HEADS * 4);
  float*  ev      = (float*) alloc((size_t)ET * HEADS * 4);
  float*  mmax    = (float*) alloc((size_t)N_NODES * HEADS * 4);
  float*  denom   = (float*) alloc((size_t)N_NODES * HEADS * 4);
  float*  agg     = (float*) alloc((size_t)N_NODES * HF * 4);    // later reused as g
  float*  deg     = (float*) alloc((size_t)N_NODES * 4);
  float*  dinv    = (float*) alloc((size_t)N_NODES * 4);
  __bf16* h_bf    = (__bf16*)alloc((size_t)N_NODES * HF * 2);
  __bf16* wgcn_t  = (__bf16*)alloc((size_t)HF * HF * 2);
  __bf16* g_bf    = (__bf16*)alloc((size_t)N_NODES * HF * 2);
  __bf16* wfc_t   = (__bf16*)alloc((size_t)HF * OUTF * 2);

  dim3 b256(256);
  auto nb = [](size_t n, int bs) { return dim3((unsigned)((n + bs - 1) / bs)); };

  // pack inputs/weights to bf16 (weights transposed to [N][K])
  cvt_f32_bf16<<<nb((size_t)N_NODES * FDIM, 256), b256, 0, stream>>>(node, node_bf, N_NODES * FDIM);
  transpose_f32_bf16<<<nb((size_t)FDIM * HF, 256), b256, 0, stream>>>(W_gat, wgat_t, FDIM, HF);
  transpose_f32_bf16<<<nb((size_t)HF * HF, 256), b256, 0, stream>>>(W_gcn, wgcn_t, HF, HF);
  transpose_f32_bf16<<<nb((size_t)HF * OUTF, 256), b256, 0, stream>>>(W_fc, wfc_t, HF, OUTF);

  // x = node @ W_gat    [10000,1280]
  gemm_bf16_wmma<<<dim3(HF / 64, N_NODES / 80), dim3(160), 0, stream>>>(
      node_bf, wgat_t, x, nullptr, N_NODES, HF, FDIM);

  // per-(node,head) attention logits
  attn_scores<<<dim3(N_NODES * HEADS / 8), b256, 0, stream>>>(x, att_src, att_dst, a_s, a_d);

  // init accumulators (every call: graph replays must be deterministic)
  fill_f32<<<nb((size_t)N_NODES * HEADS, 256), b256, 0, stream>>>(mmax, -1e30f, N_NODES * HEADS);
  fill_f32<<<nb((size_t)N_NODES * HEADS, 256), b256, 0, stream>>>(denom, 0.f, N_NODES * HEADS);
  fill_f32<<<nb((size_t)N_NODES * HF, 256), b256, 0, stream>>>(agg, 0.f, N_NODES * HF);
  fill_f32<<<nb((size_t)N_NODES, 256), b256, 0, stream>>>(deg, 0.f, N_NODES);

  // edge softmax + aggregation
  edge_scores<<<nb((size_t)ET * HEADS, 256), b256, 0, stream>>>(edge_index, a_s, a_d, ev, mmax);
  edge_exp<<<nb((size_t)ET * HEADS, 256), b256, 0, stream>>>(edge_index, ev, mmax, denom);
  edge_agg<<<dim3(ET), dim3(FDIM), 0, stream>>>(edge_index, ev, denom, x, agg, deg);

  // h = relu(agg + b_gat) -> bf16 ; degree normalization
  bias_relu_bf16<<<nb((size_t)N_NODES * HF, 256), b256, 0, stream>>>(agg, b_gat, h_bf,
                                                                     N_NODES * HF, HF);
  dinv_k<<<nb((size_t)N_NODES, 256), b256, 0, stream>>>(deg, dinv);

  // xw = h @ W_gcn   (x buffer reused)
  gemm_bf16_wmma<<<dim3(HF / 64, N_NODES / 80), dim3(160), 0, stream>>>(
      h_bf, wgcn_t, x, nullptr, N_NODES, HF, HF);

  // GCN normalized aggregation (agg buffer reused as g)
  float* g = agg;
  fill_f32<<<nb((size_t)N_NODES * HF, 256), b256, 0, stream>>>(g, 0.f, N_NODES * HF);
  gcn_agg<<<dim3(ET), dim3(FDIM), 0, stream>>>(edge_index, dinv, x, g);
  bias_relu_bf16<<<nb((size_t)N_NODES * HF, 256), b256, 0, stream>>>(g, b_gcn, g_bf,
                                                                     N_NODES * HF, HF);

  // out = g @ W_fc + b_fc   [10000,1024]
  gemm_bf16_wmma<<<dim3(OUTF / 64, N_NODES / 80), dim3(160), 0, stream>>>(
      g_bf, wfc_t, out, b_fc, N_NODES, OUTF, HF);
}
